// TransformerModel_14431090114759
// MI455X (gfx1250) — compile-verified, matrix operand order checked by reading, not run on hardware
//
#include <hip/hip_runtime.h>
#include <hip/hip_bf16.h>
#include <stdint.h>

#define DIMC   768
#define MLPC   3072
#define HEADS  12
#define HDC    64
#define DEPTH  4
#define BSZ    4
#define SEQ    1024
#define ROWS   (BSZ*SEQ)   /* 4096 */

typedef __attribute__((ext_vector_type(16))) __bf16 v16bf;
typedef __attribute__((ext_vector_type(8)))  float  v8f;

#if defined(__AMDGCN__) && __has_builtin(__builtin_amdgcn_global_load_async_to_lds_b128)
#define HAVE_ASYNC 1
#else
#define HAVE_ASYNC 0
#endif

// 16-byte int vector pointer types in explicit address spaces, matching the
// builtin's signature: (v4i AS1*, v4i AS3*, imm offset, imm cpol)
typedef int v4i_vs __attribute__((vector_size(16)));
typedef __attribute__((address_space(1))) v4i_vs glob_v4i;
typedef __attribute__((address_space(3))) v4i_vs lds_v4i;

__device__ __forceinline__ unsigned short f2bf(float f) {
  unsigned int u = __float_as_uint(f);
  unsigned int r = 0x7FFFu + ((u >> 16) & 1u);
  return (unsigned short)((u + r) >> 16);
}

// 16-byte global -> LDS copy; async (ASYNCcnt-tracked, no VGPR round trip)
// when the gfx1250 builtin is available, else synchronous fallback.
// Generic->AS3: low 32 bits of a generic LDS address are the LDS offset
// (ISA 10.2 aperture rules), so an integer truncation cast is valid.
__device__ __forceinline__ void async_cp16(const unsigned short* g, unsigned short* l) {
#if HAVE_ASYNC
  __builtin_amdgcn_global_load_async_to_lds_b128(
      (glob_v4i*)(uintptr_t)g,
      (lds_v4i*)(unsigned int)(uintptr_t)l,
      0, 0);
#else
  *(uint4*)l = *(const uint4*)g;
#endif
}

__device__ __forceinline__ void wait_async0() {
#if HAVE_ASYNC
  asm volatile("s_wait_asynccnt 0" ::: "memory");
#endif
}

union Frag16 { uint4 q[2]; v16bf v; };

// Load one 16x32 bf16 WMMA operand fragment from a row-major [rows][stride]
// ushort array (stride in elements, multiple of 8 so 16B alignment holds).
// CDNA5 16-bit A layout: lanes 0-15 -> row=lane,  K = kk+[0..7],[16..23]
//                        lanes 16-31 -> row=lane-16, K = kk+[8..15],[24..31]
// B operand uses the same layout on a [N][K]-major array (B^T rows).
__device__ __forceinline__ v16bf load_frag(const unsigned short* base, int stride,
                                           int row0, int kk, int lane) {
  const int half = lane >> 4;
  const int r = row0 + (lane & 15);
  const unsigned short* p = base + r * stride + kk + half * 8;
  Frag16 f;
  f.q[0] = *(const uint4*)(p);
  f.q[1] = *(const uint4*)(p + 16);
  return f.v;
}

__device__ __forceinline__ v8f wmma_bf16(v16bf a, v16bf b, v8f c) {
  return __builtin_amdgcn_wmma_f32_16x16x32_bf16(false, a, false, b,
                                                 (short)0, c, false, false);
}

// ---------------------------------------------------------------------------
// Weight prep: W[K][N] f32 -> Wt[N][K] bf16 (tiled 64x64 transpose via LDS)
// ---------------------------------------------------------------------------
__global__ __launch_bounds__(256)
void transpose_w_kernel(const float* __restrict__ W, unsigned short* __restrict__ Wt,
                        int K, int N) {
  __shared__ unsigned short Ts[64][72];
  const int t = threadIdx.x;
  const int n0 = blockIdx.x * 64;
  const int k0 = blockIdx.y * 64;

  const int kr = t >> 2;           // 0..63
  const int ns = (t & 3) * 16;     // 0,16,32,48
  const float* wp = W + (size_t)(k0 + kr) * N + (n0 + ns);
#pragma unroll
  for (int j = 0; j < 16; j += 4) {
    const float4 f = *(const float4*)(wp + j);
    Ts[kr][ns + j + 0] = f2bf(f.x);
    Ts[kr][ns + j + 1] = f2bf(f.y);
    Ts[kr][ns + j + 2] = f2bf(f.z);
    Ts[kr][ns + j + 3] = f2bf(f.w);
  }
  __syncthreads();

  const int nr = t >> 2;           // 0..63
  const int ks = (t & 3) * 16;     // 0,16,32,48
  unsigned short tmp[16];
#pragma unroll
  for (int j = 0; j < 16; j++) tmp[j] = Ts[ks + j][nr];
  unsigned short* op = Wt + (size_t)(n0 + nr) * K + (k0 + ks);
  *(uint4*)(op)     = *(const uint4*)(&tmp[0]);
  *(uint4*)(op + 8) = *(const uint4*)(&tmp[8]);
}

// ---------------------------------------------------------------------------
// LayerNorm: one block per row of 768, fp32 in -> bf16 out
// ---------------------------------------------------------------------------
__global__ __launch_bounds__(256)
void ln_bf16_kernel(const float* __restrict__ x, const float* __restrict__ g,
                    const float* __restrict__ b, unsigned short* __restrict__ out) {
  __shared__ float red[256];
  const int row = blockIdx.x;
  const int t = threadIdx.x;
  const float* xr = x + (size_t)row * DIMC;
  float a0 = xr[t], a1 = xr[t + 256], a2 = xr[t + 512];

  red[t] = a0 + a1 + a2;
  __syncthreads();
  for (int off = 128; off > 0; off >>= 1) {
    if (t < off) red[t] += red[t + off];
    __syncthreads();
  }
  const float mu = red[0] * (1.0f / DIMC);
  __syncthreads();

  const float d0 = a0 - mu, d1 = a1 - mu, d2 = a2 - mu;
  red[t] = d0 * d0 + d1 * d1 + d2 * d2;
  __syncthreads();
  for (int off = 128; off > 0; off >>= 1) {
    if (t < off) red[t] += red[t + off];
    __syncthreads();
  }
  const float rstd = rsqrtf(red[0] * (1.0f / DIMC) + 1e-5f);

  unsigned short* orow = out + (size_t)row * DIMC;
  orow[t]       = f2bf(d0 * rstd * g[t]       + b[t]);
  orow[t + 256] = f2bf(d1 * rstd * g[t + 256] + b[t + 256]);
  orow[t + 512] = f2bf(d2 * rstd * g[t + 512] + b[t + 512]);
}

// ---------------------------------------------------------------------------
// bf16 WMMA GEMM:  Out = epilogue(A[M,K](bf16) * W[K,N])
// BM=BN=128, BK=32, double-buffered async LDS staging; 8 waves,
// wave tile 64x32 (4x2 WMMA tiles).
// PRE=true : B from pre-transposed bf16 Wt[N][K] (pure async b128 staging)
// PRE=false: B from f32 W[K][N], converted+transposed on the fly
// EPI 0: -> bf16 (qkv) | 1: +bias,GELU -> bf16 (ff1) | 2: +bias,+resid -> f32
// ---------------------------------------------------------------------------
template <int EPI, bool PRE>
__global__ __launch_bounds__(256)
void gemm_bf16(const unsigned short* __restrict__ A,
               const float* __restrict__ Wf,
               const unsigned short* __restrict__ Wt,
               const float* __restrict__ bias,
               const float* __restrict__ resid,
               void* __restrict__ Out,
               int M, int N, int K) {
  __shared__ unsigned short As[2][128][40];   // [buf][m][k]
  __shared__ unsigned short Bs[2][128][40];   // [buf][n][k]

  const int t = threadIdx.x;
  const int lane = t & 31;
  const int w = t >> 5;
  const int wm = w >> 2;   // 0..1
  const int wn = w & 3;    // 0..3
  const int m0 = blockIdx.y * 128;
  const int n0 = blockIdx.x * 128;

  v8f zero = {0.f, 0.f, 0.f, 0.f, 0.f, 0.f, 0.f, 0.f};
  v8f acc[4][2];
#pragma unroll
  for (int i = 0; i < 4; i++)
#pragma unroll
    for (int j = 0; j < 2; j++) acc[i][j] = zero;

  // --- staging helpers -----------------------------------------------------
  auto stage = [&](int k0, int bb) {
    // A tile: 128x32 bf16, 2 x 16B per thread, async
#pragma unroll
    for (int i = 0; i < 2; i++) {
      const int c = t * 2 + i;          // 0..511
      const int row = c >> 2;
      const int ko = (c & 3) * 8;
      async_cp16(A + (size_t)(m0 + row) * K + (k0 + ko), &As[bb][row][ko]);
    }
    if (PRE) {
#pragma unroll
      for (int i = 0; i < 2; i++) {
        const int c = t * 2 + i;
        const int row = c >> 2;
        const int ko = (c & 3) * 8;
        async_cp16(Wt + (size_t)(n0 + row) * K + (k0 + ko), &Bs[bb][row][ko]);
      }
    } else {
      // f32 -> bf16 convert + transpose; pack key pairs -> b32 LDS stores
      const int kq = (t >> 4) * 2;      // 0,2,..,30
      const int cn = (t & 15) * 8;      // 0,8,..,120
      const float* w0 = Wf + (size_t)(k0 + kq) * N + (n0 + cn);
      const float* w1 = w0 + N;
      float a[8], b[8];
#pragma unroll
      for (int j = 0; j < 8; j += 4) {
        *(float4*)(&a[j]) = *(const float4*)(w0 + j);
        *(float4*)(&b[j]) = *(const float4*)(w1 + j);
      }
#pragma unroll
      for (int j = 0; j < 8; j++) {
        const unsigned int pk =
            (unsigned int)f2bf(a[j]) | ((unsigned int)f2bf(b[j]) << 16);
        *(unsigned int*)(&Bs[bb][cn + j][kq]) = pk;   // kq even -> 4B aligned
      }
    }
  };

  stage(0, 0);
  const int nst = K >> 5;
  for (int s = 0; s < nst; s++) {
    const int bb = s & 1;
    wait_async0();          // my stage-s loads have landed in LDS
    __syncthreads();        // all waves' stage-s done; buf bb^1 has no readers
    if (s + 1 < nst) stage((s + 1) << 5, bb ^ 1);   // overlap with compute

    v16bf af[4];
#pragma unroll
    for (int mt = 0; mt < 4; mt++)
      af[mt] = load_frag(&As[bb][0][0], 40, wm * 64 + mt * 16, 0, lane);
#pragma unroll
    for (int nt = 0; nt < 2; nt++) {
      v16bf bf = load_frag(&Bs[bb][0][0], 40, wn * 32 + nt * 16, 0, lane);
#pragma unroll
      for (int mt = 0; mt < 4; mt++)
        acc[mt][nt] = wmma_bf16(af[mt], bf, acc[mt][nt]);
    }
  }

  // ---- epilogue (C layout: lane<16 -> M=r, lane>=16 -> M=r+8; N=lane%16)
  const int half = lane >> 4;
  const int ln = lane & 15;
#pragma unroll
  for (int mt = 0; mt < 4; mt++) {
#pragma unroll
    for (int nt = 0; nt < 2; nt++) {
      const int col = n0 + wn * 32 + nt * 16 + ln;
#pragma unroll
      for (int r = 0; r < 8; r++) {
        const int row = m0 + wm * 64 + mt * 16 + half * 8 + r;
        float v = acc[mt][nt][r];
        if (EPI >= 1) v += bias[col];
        if (EPI == 1) v = 0.5f * v * (1.0f + erff(v * 0.70710678118654752f));
        if (EPI == 2) {
          v += resid[(size_t)row * N + col];
          ((float*)Out)[(size_t)row * N + col] = v;
        } else {
          ((unsigned short*)Out)[(size_t)row * N + col] = f2bf(v);
        }
      }
    }
  }
}

// ---------------------------------------------------------------------------
// Flash attention: one block = (b,h) x 128 query rows; 8 waves x 16 rows each.
// qkv: bf16 [4096][2304] rows laid out [q(12*64) | k(12*64) | v(12*64)]
// out: bf16 [4096][768]
// ---------------------------------------------------------------------------
__global__ __launch_bounds__(256)
void attention_kernel(const unsigned short* __restrict__ qkv,
                      unsigned short* __restrict__ out) {
  __shared__ unsigned short Qs[128][72];
  __shared__ unsigned short Ks[64][72];
  __shared__ unsigned short Vt[64][72];       // V transposed: [d][key]
  __shared__ unsigned short Ps[8][16][72];    // per-wave P tile (bf16)

  const int t = threadIdx.x;
  const int lane = t & 31;
  const int w = t >> 5;
  const int half = lane >> 4;
  const int ln = lane & 15;
  const int b = blockIdx.x / HEADS;
  const int h = blockIdx.x % HEADS;
  const int qbase = blockIdx.y * 128;
  const size_t rowbase = (size_t)b * SEQ;

  const int qoff = h * HDC;
  const int koff = DIMC + h * HDC;
  const int voff = 2 * DIMC + h * HDC;

  // load Q tile (128x64 bf16) -- async, waited inside the first kv iteration
#pragma unroll
  for (int i = 0; i < 4; i++) {
    const int idx = t * 4 + i;            // 0..1023
    const int row = idx >> 3;
    const int co = (idx & 7) * 8;
    async_cp16(qkv + (rowbase + qbase + row) * 2304 + qoff + co, &Qs[row][co]);
  }

  v8f zero = {0.f, 0.f, 0.f, 0.f, 0.f, 0.f, 0.f, 0.f};
  v8f o[4] = {zero, zero, zero, zero};
  float m_i[8], l_i[8];
#pragma unroll
  for (int r = 0; r < 8; r++) { m_i[r] = -3.0e38f; l_i[r] = 0.f; }
  const float scale = 0.125f;  // 1/sqrt(64)

  for (int kv0 = 0; kv0 < SEQ; kv0 += 64) {
    __syncthreads();   // previous readers of Ks/Vt done
    // load K chunk (64x64) -- async
#pragma unroll
    for (int i = 0; i < 2; i++) {
      const int idx = t * 2 + i;          // 0..511
      const int row = idx >> 3;
      const int co = (idx & 7) * 8;
      async_cp16(qkv + (rowbase + kv0 + row) * 2304 + koff + co, &Ks[row][co]);
    }
    // load V chunk transposed: Vt[d][key], packed pairs along key -> b32 stores
    {
      const int vk = (t >> 3) * 2;        // 0,2,..,62
      const int d0 = (t & 7) * 8;         // 0,8,..,56
      const unsigned short* vp0 = qkv + (rowbase + kv0 + vk) * 2304 + voff + d0;
      const unsigned short* vp1 = vp0 + 2304;
      unsigned short r0[8], r1[8];
      *(uint4*)(&r0[0]) = *(const uint4*)vp0;
      *(uint4*)(&r1[0]) = *(const uint4*)vp1;
#pragma unroll
      for (int j = 0; j < 8; j++) {
        const unsigned int pk = (unsigned int)r0[j] | ((unsigned int)r1[j] << 16);
        *(unsigned int*)(&Vt[d0 + j][vk]) = pk;   // vk even -> 4B aligned
      }
    }
    wait_async0();     // Q (first iter) + K chunk landed
    __syncthreads();

    // S = scale * Q K^T : this wave's 16 rows x 64 keys (4 tiles)
    float s[4][8];
#pragma unroll
    for (int nt = 0; nt < 4; nt++) {
      v8f sv = zero;
#pragma unroll
      for (int kk = 0; kk < HDC; kk += 32) {
        v16bf a = load_frag(&Qs[0][0], 72, w * 16, kk, lane);
        v16bf bb = load_frag(&Ks[0][0], 72, nt * 16, kk, lane);
        sv = wmma_bf16(a, bb, sv);
      }
#pragma unroll
      for (int r = 0; r < 8; r++) s[nt][r] = sv[r] * scale;
    }

    // online softmax (row = r + 8*half; reductions stay inside 16-lane half)
#pragma unroll
    for (int r = 0; r < 8; r++) {
      float mx = fmaxf(fmaxf(s[0][r], s[1][r]), fmaxf(s[2][r], s[3][r]));
#pragma unroll
      for (int msk = 8; msk >= 1; msk >>= 1)
        mx = fmaxf(mx, __shfl_xor(mx, msk, 32));
      const float m_new = fmaxf(m_i[r], mx);
      const float alpha = __expf(m_i[r] - m_new);
      float rs = 0.f;
#pragma unroll
      for (int nt = 0; nt < 4; nt++) {
        const float p = __expf(s[nt][r] - m_new);
        s[nt][r] = p;
        rs += p;
      }
#pragma unroll
      for (int msk = 8; msk >= 1; msk >>= 1) rs += __shfl_xor(rs, msk, 32);
      l_i[r] = l_i[r] * alpha + rs;
      m_i[r] = m_new;
#pragma unroll
      for (int dt = 0; dt < 4; dt++) o[dt][r] *= alpha;
    }

    // store P (C layout) into per-wave LDS tile as bf16 (A layout source)
#pragma unroll
    for (int nt = 0; nt < 4; nt++)
#pragma unroll
      for (int r = 0; r < 8; r++)
        Ps[w][half * 8 + r][nt * 16 + ln] = f2bf(s[nt][r]);
    // same-wave DS store -> DS load: enforce with CDNA5 split counter wait
    asm volatile("s_wait_dscnt 0" ::: "memory");

    // O += P V   (contraction over 64 keys = 2 k-steps)
#pragma unroll
    for (int kk = 0; kk < 64; kk += 32) {
      v16bf a = load_frag(&Ps[w][0][0], 72, 0, kk, lane);
#pragma unroll
      for (int dt = 0; dt < 4; dt++) {
        v16bf bb = load_frag(&Vt[0][0], 72, dt * 16, kk, lane);
        o[dt] = wmma_bf16(a, bb, o[dt]);
      }
    }
  }

  // epilogue: normalize and store bf16
#pragma unroll
  for (int dt = 0; dt < 4; dt++) {
    const int col = h * HDC + dt * 16 + ln;
#pragma unroll
    for (int r = 0; r < 8; r++) {
      const int row = qbase + w * 16 + half * 8 + r;
      out[(rowbase + row) * DIMC + col] = f2bf(o[dt][r] / l_i[r]);
    }
  }
}

// ---------------------------------------------------------------------------
extern "C" void kernel_launch(void* const* d_in, const int* in_sizes, int n_in,
                              void* d_out, int out_size, void* d_ws, size_t ws_size,
                              hipStream_t stream) {
  const float* x_in   = (const float*)d_in[0];
  const float* ln1_g  = (const float*)d_in[1];
  const float* ln1_b  = (const float*)d_in[2];
  const float* qkv_w  = (const float*)d_in[3];
  const float* proj_w = (const float*)d_in[4];
  const float* proj_b = (const float*)d_in[5];
  const float* ln2_g  = (const float*)d_in[6];
  const float* ln2_b  = (const float*)d_in[7];
  const float* ff1_w  = (const float*)d_in[8];
  const float* ff1_b  = (const float*)d_in[9];
  const float* ff2_w  = (const float*)d_in[10];
  const float* ff2_b  = (const float*)d_in[11];

  float* x = (float*)d_out;   // running residual stream lives in d_out

  char* ws = (char*)d_ws;
  unsigned short* xn   = (unsigned short*)ws;                              // [4096][768]
  unsigned short* attn = (unsigned short*)(ws + (size_t)ROWS * DIMC * 2);  // [4096][768]
  unsigned short* big  = (unsigned short*)(ws + (size_t)ROWS * DIMC * 4);  // [4096][3072]

  const size_t BASE = (size_t)ROWS * DIMC * 4 + (size_t)ROWS * MLPC * 2;   // 37,748,736
  const size_t E_QKV = (size_t)DIMC * 3 * DIMC;   // 1,769,472
  const size_t E_PRJ = (size_t)DIMC * DIMC;       //   589,824
  const size_t E_FF1 = (size_t)DIMC * MLPC;       // 2,359,296
  const size_t E_FF2 = (size_t)MLPC * DIMC;       // 2,359,296
  const size_t E_PER = E_QKV + E_PRJ + E_FF1 + E_FF2;
  const bool fast = ws_size >= BASE + E_PER * DEPTH * 2;
  unsigned short* wt0 = (unsigned short*)(ws + BASE);

  (void)hipMemcpyAsync(x, x_in, (size_t)ROWS * DIMC * sizeof(float),
                       hipMemcpyDeviceToDevice, stream);

  if (fast) {
    for (int l = 0; l < DEPTH; l++) {
      unsigned short* wl = wt0 + (size_t)l * E_PER;
      transpose_w_kernel<<<dim3(3 * DIMC / 64, DIMC / 64), 256, 0, stream>>>(
          qkv_w + (size_t)l * E_QKV, wl, DIMC, 3 * DIMC);
      transpose_w_kernel<<<dim3(DIMC / 64, DIMC / 64), 256, 0, stream>>>(
          proj_w + (size_t)l * E_PRJ, wl + E_QKV, DIMC, DIMC);
      transpose_w_kernel<<<dim3(MLPC / 64, DIMC / 64), 256, 0, stream>>>(
          ff1_w + (size_t)l * E_FF1, wl + E_QKV + E_PRJ, DIMC, MLPC);
      transpose_w_kernel<<<dim3(DIMC / 64, MLPC / 64), 256, 0, stream>>>(
          ff2_w + (size_t)l * E_FF2, wl + E_QKV + E_PRJ + E_FF1, MLPC, DIMC);
    }
  }

  for (int l = 0; l < DEPTH; l++) {
    unsigned short* wl = wt0 + (size_t)l * E_PER;
    ln_bf16_kernel<<<ROWS, 256, 0, stream>>>(x, ln1_g + l * DIMC, ln1_b + l * DIMC, xn);

    if (fast)
      gemm_bf16<0, true><<<dim3(3 * DIMC / 128, ROWS / 128), 256, 0, stream>>>(
          xn, nullptr, wl, nullptr, nullptr, big, ROWS, 3 * DIMC, DIMC);
    else
      gemm_bf16<0, false><<<dim3(3 * DIMC / 128, ROWS / 128), 256, 0, stream>>>(
          xn, qkv_w + (size_t)l * E_QKV, nullptr, nullptr, nullptr, big,
          ROWS, 3 * DIMC, DIMC);

    attention_kernel<<<dim3(BSZ * HEADS, SEQ / 128), 256, 0, stream>>>(big, attn);

    if (fast)
      gemm_bf16<2, true><<<dim3(DIMC / 128, ROWS / 128), 256, 0, stream>>>(
          attn, nullptr, wl + E_QKV, proj_b + l * DIMC, x, x, ROWS, DIMC, DIMC);
    else
      gemm_bf16<2, false><<<dim3(DIMC / 128, ROWS / 128), 256, 0, stream>>>(
          attn, proj_w + (size_t)l * E_PRJ, nullptr, proj_b + l * DIMC, x, x,
          ROWS, DIMC, DIMC);

    ln_bf16_kernel<<<ROWS, 256, 0, stream>>>(x, ln2_g + l * DIMC, ln2_b + l * DIMC, xn);

    if (fast)
      gemm_bf16<1, true><<<dim3(MLPC / 128, ROWS / 128), 256, 0, stream>>>(
          xn, nullptr, wl + E_QKV + E_PRJ, ff1_b + l * MLPC, nullptr, big,
          ROWS, MLPC, DIMC);
    else
      gemm_bf16<1, false><<<dim3(MLPC / 128, ROWS / 128), 256, 0, stream>>>(
          xn, ff1_w + (size_t)l * E_FF1, nullptr, ff1_b + l * MLPC, nullptr, big,
          ROWS, MLPC, DIMC);

    if (fast)
      gemm_bf16<2, true><<<dim3(DIMC / 128, ROWS / 128), 256, 0, stream>>>(
          big, nullptr, wl + E_QKV + E_PRJ + E_FF1, ff2_b + l * DIMC, x, x,
          ROWS, DIMC, MLPC);
    else
      gemm_bf16<2, false><<<dim3(DIMC / 128, ROWS / 128), 256, 0, stream>>>(
          big, ff2_w + (size_t)l * E_FF2, nullptr, ff2_b + l * DIMC, x, x,
          ROWS, DIMC, MLPC);
  }
}
